// ResBlock_57904749085416
// MI455X (gfx1250) — matrix operand order repaired and checked by reading, not compiled, not run on hardware
//
#include <hip/hip_runtime.h>
#include <hip/hip_bf16.h>

// ---------------------------------------------------------------------------
// HEALPix Swin block for MI455X (gfx1250, wave32, WMMA + Tensor Data Mover).
// fp32 master activations; bf16 WMMA (v_wmma_f32_16x16x32_bf16) for all GEMMs;
// TDM (tensor_load_to_lds) streams A/B tiles into LDS, double-buffered on
// TENSORcnt, so VALU/VMEM pipes are free for compute on this HBM-bound op.
// ---------------------------------------------------------------------------

#define NSIDE 128
#define NPIX (12 * NSIDE * NSIDE)   // 196608 tokens
#define FDIM 64
#define CDIM 128
#define MDIM 512

typedef __bf16 bf16;
typedef __attribute__((ext_vector_type(16))) __bf16 v16bf;
typedef __attribute__((ext_vector_type(8)))  float  v8f;
typedef unsigned int v4u  __attribute__((ext_vector_type(4)));
typedef int          v8i_t __attribute__((ext_vector_type(8)));
typedef int          v4i_t __attribute__((ext_vector_type(4)));

__device__ __forceinline__ float gelu_exact(float x) {
    return 0.5f * x * (1.0f + erff(x * 0.7071067811865476f));
}

// ---------------------------------------------------------------------------
// TDM: issue a 2D tile load Global->LDS (D# per CDNA5 ISA).
//   dsz: log2 element bytes (1=2B,2=4B). td0 elements/row, td1 rows.
//   stride0: row stride in elements. Pad: +1 DWORD after every 32 DWORDs (128B),
//   so a 128B tile row lands with an odd-DWORD LDS stride (bank stagger).
// ---------------------------------------------------------------------------
__device__ __forceinline__ void tdm_load_2d(unsigned lds_addr, const void* gptr,
                                            unsigned dsz, unsigned td0, unsigned td1,
                                            unsigned long long stride0) {
    unsigned long long ga = (unsigned long long)(uintptr_t)gptr;
    v4u g0;
    g0[0] = 1u;                                                  // count=1, user mode
    g0[1] = lds_addr;                                            // LDS byte address
    g0[2] = (unsigned)ga;                                        // global_addr[31:0]
    g0[3] = (unsigned)((ga >> 32) & 0x01FFFFFFu) | (2u << 30);   // addr[56:32] | type=2
    v8i_t g1;
    g1[0] = (int)((dsz << 16) | (1u << 20) | (4u << 22));        // data_size | pad_en | interval=32dw (pad=1dw)
    g1[1] = (int)((td0 & 0xFFFFu) << 16);                        // tensor_dim0[15:0] (== tile: in-bounds)
    g1[2] = (int)(((td0 >> 16) & 0xFFFFu) | ((td1 & 0xFFFFu) << 16)); // tdim0 hi | tdim1 lo
    g1[3] = (int)(((td1 >> 16) & 0xFFFFu) | (td0 << 16));        // tdim1 hi | tile_dim0
    g1[4] = (int)(td1 & 0xFFFFu);                                // tile_dim1 (tile_dim2=0 -> 2D)
    g1[5] = (int)(unsigned)(stride0 & 0xFFFFFFFFull);            // tensor_dim0_stride lo
    g1[6] = (int)(unsigned)((stride0 >> 32) & 0xFFFFull);        // stride hi | dim1_stride=0
    g1[7] = 0;
    v4i_t gz = {0, 0, 0, 0};
#if __clang_major__ >= 23
    v8i_t gz8 = {0, 0, 0, 0, 0, 0, 0, 0};
    __builtin_amdgcn_tensor_load_to_lds(g0, g1, gz, gz, gz8, 0);
#else
    __builtin_amdgcn_tensor_load_to_lds(g0, g1, gz, gz, 0);
#endif
}

// ---------------------------------------------------------------------------
// fp32 -> bf16 weight conversion (weights are tiny; L2-resident afterwards)
// ---------------------------------------------------------------------------
__global__ void cvt_bf16_kernel(const float* __restrict__ src,
                                bf16* __restrict__ dst, int n) {
    int i = blockIdx.x * 256 + threadIdx.x;
    if (i < n) dst[i] = (bf16)src[i];
}

// ---------------------------------------------------------------------------
// LayerNorm over C=128: one wave (32 lanes) per token, 4 channels per lane.
// ---------------------------------------------------------------------------
__global__ __launch_bounds__(256)
void layernorm_kernel(const float* __restrict__ x, const float* __restrict__ g,
                      const float* __restrict__ b, float* __restrict__ y, int ntok) {
    int wave = threadIdx.x >> 5;
    int lane = threadIdx.x & 31;
    int tok  = blockIdx.x * 8 + wave;
    if (tok >= ntok) return;
    const float4 v = *(const float4*)(x + (size_t)tok * CDIM + lane * 4);
    float s = v.x + v.y + v.z + v.w;
#pragma unroll
    for (int m = 16; m; m >>= 1) s += __shfl_xor(s, m, 32);
    float mu = s * (1.0f / (float)CDIM);
    float dx = v.x - mu, dy = v.y - mu, dz = v.z - mu, dw = v.w - mu;
    float q = dx * dx + dy * dy + dz * dz + dw * dw;
#pragma unroll
    for (int m = 16; m; m >>= 1) q += __shfl_xor(q, m, 32);
    float inv = rsqrtf(q * (1.0f / (float)CDIM) + 1e-5f);
    const float4 gg = *(const float4*)(g + lane * 4);
    const float4 bb = *(const float4*)(b + lane * 4);
    float4 o;
    o.x = dx * inv * gg.x + bb.x;
    o.y = dy * inv * gg.y + bb.y;
    o.z = dz * inv * gg.z + bb.z;
    o.w = dw * inv * gg.w + bb.w;
    *(float4*)(y + (size_t)tok * CDIM + lane * 4) = o;
}

// ---------------------------------------------------------------------------
// Tiled GEMM with TDM-streamed tiles:
//   C[M,N] = epilogue(A[M,K] @ B[K,N] + bias[N]) (+ residual)
//   Block = 128 threads (4 waves), tile = 64(M) x 64(N), K step 32.
//   Wave w owns rows [w*16, w*16+16): 1 A-fragment, 4 WMMAs per K-step.
//   A fp32 tile (64x32, LDS stride 33 via TDM pad) converted bf16 at pack time;
//   B bf16 tile (32x64, LDS stride 66). Double-buffered on TENSORcnt.
//   EPI: 0 = bias, 1 = bias+GELU, 2 = bias+residual (res ldc == Ncols)
// ---------------------------------------------------------------------------
template <int EPI>
__global__ __launch_bounds__(128)
void gemm_tdm_kernel(const float* __restrict__ A, const bf16* __restrict__ B,
                     const float* __restrict__ bias, const float* __restrict__ res,
                     float* __restrict__ C, int K, int Ncols) {
    __shared__ alignas(16) float lAf[2][64 * 33];   // 16896 B
    __shared__ alignas(16) bf16  lBh[2][32 * 66];   //  8448 B

    const int tid  = threadIdx.x;
    const int lane = tid & 31;
    const int wave = tid >> 5;
    const int half = lane >> 4;
    const int l16  = lane & 15;
    const int m0   = blockIdx.y * 64;
    const int n0   = blockIdx.x * 64;
    const int ksteps = K >> 5;

    if (wave == 0) {
        tdm_load_2d((unsigned)(uintptr_t)&lAf[0][0], A + (size_t)m0 * K,
                    2, 32, 64, (unsigned long long)K);
        tdm_load_2d((unsigned)(uintptr_t)&lBh[0][0], B + (size_t)n0,
                    1, 64, 32, (unsigned long long)Ncols);
    }

    v8f acc[4] = {v8f{}, v8f{}, v8f{}, v8f{}};

    for (int s = 0; s < ksteps; ++s) {
        const int buf = s & 1;
        if (wave == 0) {
            if (s + 1 < ksteps) {
                const int k1 = (s + 1) << 5;
                tdm_load_2d((unsigned)(uintptr_t)&lAf[buf ^ 1][0],
                            A + (size_t)m0 * K + k1, 2, 32, 64, (unsigned long long)K);
                tdm_load_2d((unsigned)(uintptr_t)&lBh[buf ^ 1][0],
                            B + (size_t)k1 * Ncols + n0, 1, 64, 32, (unsigned long long)Ncols);
                __builtin_amdgcn_s_wait_tensorcnt((short)2);   // oldest pair (this step) done
            } else {
                __builtin_amdgcn_s_wait_tensorcnt((short)0);
            }
            __asm__ volatile("" ::: "memory");                 // LDS written by TDM, not by IR stores
        }
        __syncthreads();

        // Pack A fragment (ISA 7.12.2 16-bit A 16x32): row = lane%16 of wave slice.
        v16bf afrag;
        const float* ar = &lAf[buf][(wave * 16 + l16) * 33];
#pragma unroll
        for (int j = 0; j < 8; ++j) {
            int ka = (j < 4) ? (half * 8 + 2 * j) : (16 + half * 8 + 2 * (j - 4));
            afrag[2 * j]     = (bf16)ar[ka];
            afrag[2 * j + 1] = (bf16)ar[ka + 1];
        }
#pragma unroll
        for (int nt = 0; nt < 4; ++nt) {
            v16bf bfrag;
#pragma unroll
            for (int j = 0; j < 8; ++j) {
                int kb = (j < 4) ? (half * 8 + 2 * j) : (16 + half * 8 + 2 * (j - 4));
                bfrag[2 * j]     = lBh[buf][kb * 66 + nt * 16 + l16];
                bfrag[2 * j + 1] = lBh[buf][(kb + 1) * 66 + nt * 16 + l16];
            }
            acc[nt] = __builtin_amdgcn_wmma_f32_16x16x32_bf16(
                false, afrag, false, bfrag, (short)0, acc[nt], false, false);
        }
        __syncthreads();
    }

    // Epilogue: C layout — VGPR r holds M = r + 8*half, N = lane%16.
    const int mrow = m0 + wave * 16;
#pragma unroll
    for (int nt = 0; nt < 4; ++nt) {
        const int n = n0 + nt * 16 + l16;
        const float bv = bias[n];
#pragma unroll
        for (int r = 0; r < 8; ++r) {
            int m = mrow + r + 8 * half;
            float v = acc[nt][r] + bv;
            if (EPI == 1) v = gelu_exact(v);
            if (EPI == 2) v += res[(size_t)m * Ncols + n];
            C[(size_t)m * Ncols + n] = v;
        }
    }
}

// ---------------------------------------------------------------------------
// Window attention: one block per 16-token window, one wave per head (H=4,D=32).
// qkv[N,384] = [q|k|v] x [H=4][D=32]. Cyclic shift folded into window indexing.
//   scores = Q(16x32) @ K^T(32x16) via one WMMA;  softmax via LDS (intra-wave);
//   out    = P(16x16, K-padded) @ V(16x32) via two WMMAs.
// ---------------------------------------------------------------------------
__global__ __launch_bounds__(128)
void window_attn_kernel(const float* __restrict__ qkv, float* __restrict__ attn,
                        int ntok, int shift) {
    __shared__ float sc[4][16][17];
    const int lane = threadIdx.x & 31;
    const int head = threadIdx.x >> 5;
    const int half = lane >> 4;
    const int l16  = lane & 15;
    const int base = blockIdx.x * 16 + shift;

    int tq = base + l16; if (tq >= ntok) tq -= ntok;   // query row token == key col token
    const float* qp = qkv + (size_t)tq * 384 + head * 32;          // Q
    const float* kp = qkv + (size_t)tq * 384 + CDIM + head * 32;   // K

    v16bf qf, kf;
#pragma unroll
    for (int j = 0; j < 8; ++j) {
        int ka = (j < 4) ? (half * 8 + 2 * j) : (16 + half * 8 + 2 * (j - 4));
        qf[2 * j]     = (bf16)qp[ka];
        qf[2 * j + 1] = (bf16)qp[ka + 1];
        int kd = (j < 4) ? (half * 8 + 2 * j) : (16 + half * 8 + 2 * (j - 4));
        kf[2 * j]     = (bf16)kp[kd];
        kf[2 * j + 1] = (bf16)kp[kd + 1];
    }
    v8f s = {};
    s = __builtin_amdgcn_wmma_f32_16x16x32_bf16(
        false, qf, false, kf, (short)0, s, false, false);

    const float scale = 0.17677669529663687f;   // 32^-0.5
#pragma unroll
    for (int r = 0; r < 8; ++r) sc[head][r + 8 * half][l16] = s[r] * scale;

    // Softmax per row; lanes 0..15 each own one query row (intra-wave LDS order).
    if (half == 0) {
        float mx = -3.0e38f;
#pragma unroll
        for (int c = 0; c < 16; ++c) mx = fmaxf(mx, sc[head][l16][c]);
        float sum = 0.0f;
#pragma unroll
        for (int c = 0; c < 16; ++c) {
            float e = __expf(sc[head][l16][c] - mx);
            sc[head][l16][c] = e;
            sum += e;
        }
        float rn = 1.0f / sum;
#pragma unroll
        for (int c = 0; c < 16; ++c) sc[head][l16][c] *= rn;
    }

    // out = P @ V, two 16-col halves of D=32; K=16 keys zero-padded to 32.
#pragma unroll
    for (int hh = 0; hh < 2; ++hh) {
        v16bf pf, vf;
#pragma unroll
        for (int j = 0; j < 8; ++j) {
            if (j < 4) {
                int ka = half * 8 + 2 * j;                 // K < 16: real probs
                pf[2 * j]     = (bf16)sc[head][l16][ka];
                pf[2 * j + 1] = (bf16)sc[head][l16][ka + 1];
            } else {
                pf[2 * j] = (bf16)0.0f; pf[2 * j + 1] = (bf16)0.0f;
            }
            if (j < 4) {                                   // K = half*8 + 2j (+1) < 16
                int t0 = base + half * 8 + 2 * j;     if (t0 >= ntok) t0 -= ntok;
                int t1 = base + half * 8 + 2 * j + 1; if (t1 >= ntok) t1 -= ntok;
                vf[2 * j]     = (bf16)qkv[(size_t)t0 * 384 + 2 * CDIM + head * 32 + hh * 16 + l16];
                vf[2 * j + 1] = (bf16)qkv[(size_t)t1 * 384 + 2 * CDIM + head * 32 + hh * 16 + l16];
            } else {                                       // K=16..31: padding
                vf[2 * j] = (bf16)0.0f; vf[2 * j + 1] = (bf16)0.0f;
            }
        }
        v8f o = {};
        o = __builtin_amdgcn_wmma_f32_16x16x32_bf16(
            false, pf, false, vf, (short)0, o, false, false);
#pragma unroll
        for (int r = 0; r < 8; ++r) {
            int m = r + 8 * half;
            int tm = base + m; if (tm >= ntok) tm -= ntok;
            attn[(size_t)tm * CDIM + head * 32 + hh * 16 + l16] = o[r];
        }
    }
}

// ---------------------------------------------------------------------------
// Host-side orchestration
// ---------------------------------------------------------------------------
extern "C" void kernel_launch(void* const* d_in, const int* in_sizes, int n_in,
                              void* d_out, int out_size, void* d_ws, size_t ws_size,
                              hipStream_t stream) {
    (void)in_sizes; (void)n_in; (void)out_size; (void)ws_size;

    const float* x_in   = (const float*)d_in[0];   // [N,64]
    const float* w_ff1  = (const float*)d_in[1];   // [64,128]
    const float* b_ff1  = (const float*)d_in[2];
    const float* ln1_g  = (const float*)d_in[3];   // [2,128]
    const float* ln1_b  = (const float*)d_in[4];
    const float* w_qkv  = (const float*)d_in[5];   // [2,128,384]
    const float* b_qkv  = (const float*)d_in[6];   // [2,384]
    const float* w_proj = (const float*)d_in[7];   // [2,128,128]
    const float* b_proj = (const float*)d_in[8];
    const float* ln2_g  = (const float*)d_in[9];
    const float* ln2_b  = (const float*)d_in[10];
    const float* w_mlp1 = (const float*)d_in[11];  // [2,128,512]
    const float* b_mlp1 = (const float*)d_in[12];
    const float* w_mlp2 = (const float*)d_in[13];  // [2,512,128]
    const float* b_mlp2 = (const float*)d_in[14];
    const float* w_ff2  = (const float*)d_in[15];  // [128,64]
    const float* b_ff2  = (const float*)d_in[16];

    char* ws = (char*)d_ws;
    // bf16 weight regions (all 256B-aligned)
    bf16* WFF1  = (bf16*)(ws + 0);          // 64*128
    bf16* WQKV  = (bf16*)(ws + 16384);      // 2*128*384
    bf16* WPROJ = (bf16*)(ws + 212992);     // 2*128*128
    bf16* WMLP1 = (bf16*)(ws + 278528);     // 2*128*512
    bf16* WMLP2 = (bf16*)(ws + 540672);     // 2*512*128
    bf16* WFF2  = (bf16*)(ws + 802816);     // 128*64
    // fp32 activation buffers
    const size_t XB = (size_t)NPIX * CDIM * sizeof(float);  // 100663296
    float* xbuf = (float*)(ws + (1u << 20));
    float* tbuf = (float*)(ws + (1u << 20) + XB);
    float* big  = (float*)(ws + (1u << 20) + 2 * XB);       // qkv[N,384] / mlp1[N,512]

    auto cvt = [&](const float* s, bf16* d, int n) {
        cvt_bf16_kernel<<<(n + 255) / 256, 256, 0, stream>>>(s, d, n);
    };
    cvt(w_ff1,  WFF1,  FDIM * CDIM);
    cvt(w_qkv,  WQKV,  2 * CDIM * 3 * CDIM);
    cvt(w_proj, WPROJ, 2 * CDIM * CDIM);
    cvt(w_mlp1, WMLP1, 2 * CDIM * MDIM);
    cvt(w_mlp2, WMLP2, 2 * MDIM * CDIM);
    cvt(w_ff2,  WFF2,  CDIM * FDIM);

    const dim3 blk(128);
    const int MTILES = NPIX / 64;  // 3072

    // h = gelu(x @ w_ff1 + b_ff1)
    gemm_tdm_kernel<1><<<dim3(CDIM / 64, MTILES), blk, 0, stream>>>(
        x_in, WFF1, b_ff1, nullptr, xbuf, FDIM, CDIM);

    for (int i = 0; i < 2; ++i) {
        const int shift = i ? (16 / 2) : 0;
        // --- attention sub-block ---
        layernorm_kernel<<<NPIX / 8, 256, 0, stream>>>(
            xbuf, ln1_g + i * CDIM, ln1_b + i * CDIM, tbuf, NPIX);
        gemm_tdm_kernel<0><<<dim3(3 * CDIM / 64, MTILES), blk, 0, stream>>>(
            tbuf, WQKV + (size_t)i * CDIM * 3 * CDIM, b_qkv + i * 3 * CDIM,
            nullptr, big, CDIM, 3 * CDIM);
        window_attn_kernel<<<NPIX / 16, blk, 0, stream>>>(big, tbuf, NPIX, shift);
        gemm_tdm_kernel<2><<<dim3(CDIM / 64, MTILES), blk, 0, stream>>>(
            tbuf, WPROJ + (size_t)i * CDIM * CDIM, b_proj + i * CDIM,
            xbuf, xbuf, CDIM, CDIM);
        // --- MLP sub-block ---
        layernorm_kernel<<<NPIX / 8, 256, 0, stream>>>(
            xbuf, ln2_g + i * CDIM, ln2_b + i * CDIM, tbuf, NPIX);
        gemm_tdm_kernel<1><<<dim3(MDIM / 64, MTILES), blk, 0, stream>>>(
            tbuf, WMLP1 + (size_t)i * CDIM * MDIM, b_mlp1 + i * MDIM,
            nullptr, big, CDIM, MDIM);
        gemm_tdm_kernel<2><<<dim3(CDIM / 64, MTILES), blk, 0, stream>>>(
            big, WMLP2 + (size_t)i * MDIM * CDIM, b_mlp2 + i * CDIM,
            xbuf, xbuf, MDIM, CDIM);
    }

    // out = x_in + h @ w_ff2 + b_ff2
    gemm_tdm_kernel<2><<<dim3(FDIM / 64, MTILES), blk, 0, stream>>>(
        xbuf, WFF2, b_ff2, x_in, (float*)d_out, CDIM, FDIM);
}
